// PositionwiseFeedForwardMoE_29841432773030
// MI455X (gfx1250) — compile-verified, matrix operand order checked by reading, not run on hardware
//
#include <hip/hip_runtime.h>
#include <hip/hip_bf16.h>
#include <stdint.h>

#define N_TOK 8192
#define D_DIM 1024
#define E_EXP 8
#define H_DIM 1024
#define TILE_M 32

typedef __attribute__((ext_vector_type(16))) __bf16 v16bf;
typedef __attribute__((ext_vector_type(8)))  float  v8f;
typedef unsigned int u32x4 __attribute__((ext_vector_type(4)));
typedef int          i32x4 __attribute__((ext_vector_type(4)));
typedef int          i32x8 __attribute__((ext_vector_type(8)));

#if defined(__has_builtin)
#if __has_builtin(__builtin_amdgcn_tensor_load_to_lds) && __has_builtin(__builtin_amdgcn_s_wait_tensorcnt)
#define HAVE_TDM 1
#endif
#endif
#ifndef HAVE_TDM
#define HAVE_TDM 0
#endif

union FragA { v16bf v; uint4 q[2]; };

// ---------------------------------------------------------------------------
// f32 -> bf16 straight convert (for x)
// ---------------------------------------------------------------------------
__global__ void cvt_x_kernel(const float* __restrict__ x, __bf16* __restrict__ xb, int n) {
  for (int i = blockIdx.x * blockDim.x + threadIdx.x; i < n; i += gridDim.x * blockDim.x)
    xb[i] = (__bf16)x[i];
}

// ---------------------------------------------------------------------------
// per-expert transpose+convert  w[e][K][Nn] (f32) -> wt[e][Nn][K] (bf16)
// ---------------------------------------------------------------------------
__global__ void cvt_wT_kernel(const float* __restrict__ w, __bf16* __restrict__ wt,
                              int K, int Nn) {
  const float* wsrc = w  + (size_t)blockIdx.y * K * Nn;
  __bf16*      wdst = wt + (size_t)blockIdx.y * K * Nn;
  const int total = K * Nn;
  for (int i = blockIdx.x * blockDim.x + threadIdx.x; i < total; i += gridDim.x * blockDim.x) {
    int k = i % K;
    int n = i / K;
    wdst[i] = (__bf16)wsrc[(size_t)k * Nn + n];
  }
}

// ---------------------------------------------------------------------------
// Router: one wave32 per token.
// ---------------------------------------------------------------------------
__global__ void router_kernel(const float* __restrict__ x, const float* __restrict__ Wp,
                              const float* __restrict__ bp,
                              float* __restrict__ probs_out, int* __restrict__ chosen_out,
                              int* __restrict__ counts, int* __restrict__ lists) {
  const int wave  = threadIdx.x >> 5;
  const int lane  = threadIdx.x & 31;
  const int token = blockIdx.x * (blockDim.x >> 5) + wave;
  if (token >= N_TOK) return;

  float acc[E_EXP];
#pragma unroll
  for (int e = 0; e < E_EXP; ++e) acc[e] = 0.f;

  const float* xr = x + (size_t)token * D_DIM;
  for (int d = lane; d < D_DIM; d += 32) {
    const float xv = xr[d];
    const float* wr = Wp + (size_t)d * E_EXP;
#pragma unroll
    for (int e = 0; e < E_EXP; ++e) acc[e] += xv * wr[e];
  }
#pragma unroll
  for (int e = 0; e < E_EXP; ++e) {
#pragma unroll
    for (int m = 16; m >= 1; m >>= 1) acc[e] += __shfl_xor(acc[e], m, 32);
  }

  if (lane == 0) {
    float mx = -3.0e38f;
    int arg = 0;
#pragma unroll
    for (int e = 0; e < E_EXP; ++e) {
      acc[e] += bp[e];
      if (acc[e] > mx) { mx = acc[e]; arg = e; }
    }
    float s = 0.f;
#pragma unroll
    for (int e = 0; e < E_EXP; ++e) { acc[e] = __expf(acc[e] - mx); s += acc[e]; }
    const float inv = 1.f / s;
#pragma unroll
    for (int e = 0; e < E_EXP; ++e) probs_out[(size_t)token * E_EXP + e] = acc[e] * inv;
    chosen_out[token] = arg;
    const int pos = atomicAdd(&counts[arg], 1);
    lists[arg * N_TOK + pos] = token;
  }
}

// ---------------------------------------------------------------------------
// Expert FFN: grid = (tiles, experts), 256 threads = 8 wave32.
// X-tile gather done by the Tensor Data Mover in gather mode (2 descriptors
// of 16 row-indices each), tracked with TENSORcnt.
// ---------------------------------------------------------------------------
__global__ void __launch_bounds__(256, 1)
moe_ffn_kernel(const __bf16* __restrict__ Xb, const __bf16* __restrict__ W1t,
               const __bf16* __restrict__ W2t, const float* __restrict__ b1,
               const float* __restrict__ b2, const int* __restrict__ counts,
               const int* __restrict__ lists, float* __restrict__ out) {
  const int e    = blockIdx.y;
  const int tile = blockIdx.x;
  const int cnt  = counts[e];
  if (tile * TILE_M >= cnt) return;

  const int* lst = lists + e * N_TOK;

  extern __shared__ __bf16 smem[];
  __bf16* Xs = smem;                       // [TILE_M][D_DIM]  64 KB
  __bf16* Hs = smem + TILE_M * D_DIM;      // [TILE_M][H_DIM]  64 KB
  __shared__ int toks[TILE_M];

  const int tid = threadIdx.x;
  if (tid < TILE_M) {
    int i = tile * TILE_M + tid;
    if (i > cnt - 1) i = cnt - 1;          // clamp: duplicate compute, masked store
    toks[tid] = lst[i];
  }

#if HAVE_TDM
  // -------- TDM gather: 2 descriptors x 16 token rows -> Xs (wave 0) -------
  if (tid < 32) {
    const uint32_t ldsbase = (uint32_t)(uintptr_t)(void*)Xs;
    const uint64_t gbase   = (uint64_t)(uintptr_t)Xb;
#pragma unroll 1
    for (int half = 0; half < 2; ++half) {
      // D# group 0: count=1, gather_mode=1, 16-bit indices, lds_addr, global_addr, type=2
      u32x4 g0;
      g0[0] = 0x80000001u;
      g0[1] = ldsbase + (uint32_t)half * (16u * D_DIM * 2u);
      g0[2] = (uint32_t)gbase;
      g0[3] = (uint32_t)((gbase >> 32) & 0x01FFFFFFull) | (2u << 30);
      // D# group 1: data_size=2B, tensor_dim0=1024, tensor_dim1=8192,
      // tile_dim0=1024, tile_dim1=16 indices, tensor_dim0_stride=1024
      i32x8 g1;
      g1[0] = (int)(1u << 16);                                      // data_size=1 (2B)
      g1[1] = (int)((uint32_t)(D_DIM & 0xFFFF) << 16);              // dim0 lo16
      g1[2] = (int)(((uint32_t)(N_TOK & 0xFFFF) << 16) |
                    ((uint32_t)D_DIM >> 16));                       // dim1 lo16 | dim0 hi16
      g1[3] = (int)(((uint32_t)(D_DIM & 0xFFFF) << 16) |
                    ((uint32_t)N_TOK >> 16));                       // tile_dim0 | dim1 hi16
      g1[4] = 16;                                                   // tile_dim1 = #indices
      g1[5] = D_DIM;                                                // dim0_stride lo32
      g1[6] = 0;
      g1[7] = 0;
      // D# groups 2+3: 16 row indices (token ids), 16-bit packed
      int idxdw[8];
#pragma unroll
      for (int j = 0; j < 8; ++j) {
        const int t0 = toks[half * 16 + 2 * j];
        const int t1 = toks[half * 16 + 2 * j + 1];
        idxdw[j] = (t0 & 0xFFFF) | (t1 << 16);
      }
      i32x4 g2, g3;
#pragma unroll
      for (int j = 0; j < 4; ++j) { g2[j] = idxdw[j]; g3[j] = idxdw[4 + j]; }
#if __clang_major__ >= 23
      i32x8 z8 = {};
      __builtin_amdgcn_tensor_load_to_lds(g0, g1, g2, g3, z8, 0);
#else
      __builtin_amdgcn_tensor_load_to_lds(g0, g1, g2, g3, 0);
#endif
    }
    __builtin_amdgcn_s_wait_tensorcnt(0);
  }
  __syncthreads();
#else
  __syncthreads();
  {
    const int vecPerRow = D_DIM / 8;       // 128 x uint4 per row
    for (int i = tid; i < TILE_M * vecPerRow; i += 256) {
      const int r = i / vecPerRow, c = i % vecPerRow;
      ((uint4*)(Xs + r * D_DIM))[c] = ((const uint4*)(Xb + (size_t)toks[r] * D_DIM))[c];
    }
  }
  __syncthreads();
#endif

  const int wave    = tid >> 5;
  const int lane    = tid & 31;
  const int halfsel = (lane < 16) ? 0 : 8;   // K-chunk select per WMMA A/B layout
  const int nlane   = lane & 15;

  // ---------------- stage 1: Hs = relu(Xs @ W1 + b1) --------------------
  const __bf16* w1e = W1t + (size_t)e * H_DIM * D_DIM;   // [H][D]
#pragma unroll 1
  for (int nt = 0; nt < 8; ++nt) {
    const int col = (wave * 8 + nt) * 16 + nlane;        // output column in H
    v8f acc0 = {}; v8f acc1 = {};
    const __bf16* brow = w1e + (size_t)col * D_DIM;
    const __bf16* xr0  = Xs + (size_t)nlane * D_DIM + halfsel;
    const __bf16* xr1  = Xs + (size_t)(16 + nlane) * D_DIM + halfsel;
#pragma unroll 1
    for (int ks = 0; ks < D_DIM / 32; ++ks) {
      const int k0 = ks * 32;
      FragA a0, a1, b;
      a0.q[0] = *(const uint4*)(xr0 + k0);
      a0.q[1] = *(const uint4*)(xr0 + k0 + 16);
      a1.q[0] = *(const uint4*)(xr1 + k0);
      a1.q[1] = *(const uint4*)(xr1 + k0 + 16);
      b.q[0]  = *(const uint4*)(brow + k0 + halfsel);
      b.q[1]  = *(const uint4*)(brow + k0 + halfsel + 16);
      __builtin_prefetch(brow + k0 + 256, 0, 1);
      acc0 = __builtin_amdgcn_wmma_f32_16x16x32_bf16(false, a0.v, false, b.v,
                                                     (short)0, acc0, false, false);
      acc1 = __builtin_amdgcn_wmma_f32_16x16x32_bf16(false, a1.v, false, b.v,
                                                     (short)0, acc1, false, false);
    }
    const float bias = b1[e * H_DIM + col];
#pragma unroll
    for (int j = 0; j < 8; ++j) {
      float h0 = acc0[j] + bias; h0 = h0 > 0.f ? h0 : 0.f;
      float h1 = acc1[j] + bias; h1 = h1 > 0.f ? h1 : 0.f;
      Hs[(size_t)(j + halfsel) * H_DIM + col]      = (__bf16)h0;
      Hs[(size_t)(16 + j + halfsel) * H_DIM + col] = (__bf16)h1;
    }
  }
  __syncthreads();

  // ---------------- stage 2: out = Hs @ W2 + b2 -------------------------
  const __bf16* w2e = W2t + (size_t)e * D_DIM * H_DIM;   // [D][H]
#pragma unroll 1
  for (int nt = 0; nt < 8; ++nt) {
    const int col = (wave * 8 + nt) * 16 + nlane;        // output column in D
    v8f acc0 = {}; v8f acc1 = {};
    const __bf16* brow = w2e + (size_t)col * H_DIM;
    const __bf16* hr0  = Hs + (size_t)nlane * H_DIM + halfsel;
    const __bf16* hr1  = Hs + (size_t)(16 + nlane) * H_DIM + halfsel;
#pragma unroll 1
    for (int ks = 0; ks < H_DIM / 32; ++ks) {
      const int k0 = ks * 32;
      FragA a0, a1, b;
      a0.q[0] = *(const uint4*)(hr0 + k0);
      a0.q[1] = *(const uint4*)(hr0 + k0 + 16);
      a1.q[0] = *(const uint4*)(hr1 + k0);
      a1.q[1] = *(const uint4*)(hr1 + k0 + 16);
      b.q[0]  = *(const uint4*)(brow + k0 + halfsel);
      b.q[1]  = *(const uint4*)(brow + k0 + halfsel + 16);
      __builtin_prefetch(brow + k0 + 256, 0, 1);
      acc0 = __builtin_amdgcn_wmma_f32_16x16x32_bf16(false, a0.v, false, b.v,
                                                     (short)0, acc0, false, false);
      acc1 = __builtin_amdgcn_wmma_f32_16x16x32_bf16(false, a1.v, false, b.v,
                                                     (short)0, acc1, false, false);
    }
    const float bias = b2[e * D_DIM + col];
#pragma unroll
    for (int j = 0; j < 8; ++j) {
      int r = j + halfsel;
      if (tile * TILE_M + r < cnt)
        out[(size_t)toks[r] * D_DIM + col] = acc0[j] + bias;
      r = 16 + j + halfsel;
      if (tile * TILE_M + r < cnt)
        out[(size_t)toks[r] * D_DIM + col] = acc1[j] + bias;
    }
  }
}

// ---------------------------------------------------------------------------
extern "C" void kernel_launch(void* const* d_in, const int* in_sizes, int n_in,
                              void* d_out, int out_size, void* d_ws, size_t ws_size,
                              hipStream_t stream) {
  (void)in_sizes; (void)n_in; (void)out_size; (void)ws_size;

  const float* x  = (const float*)d_in[0];
  const float* Wp = (const float*)d_in[1];
  const float* bp = (const float*)d_in[2];
  const float* W1 = (const float*)d_in[3];
  const float* b1 = (const float*)d_in[4];
  const float* W2 = (const float*)d_in[5];
  const float* b2 = (const float*)d_in[6];

  float* out    = (float*)d_out;                                   // [N, D]
  float* probs  = out + (size_t)N_TOK * D_DIM;                     // [N, E]
  int*   chosen = (int*)(probs + (size_t)N_TOK * E_EXP);           // [N]

  char* ws = (char*)d_ws;
  const size_t wbytes = (size_t)E_EXP * D_DIM * H_DIM * sizeof(__bf16);  // 16 MB
  __bf16* W1t    = (__bf16*)(ws);
  __bf16* W2t    = (__bf16*)(ws + wbytes);
  __bf16* Xb     = (__bf16*)(ws + 2 * wbytes);
  int*    counts = (int*)(ws + 2 * wbytes + (size_t)N_TOK * D_DIM * sizeof(__bf16));
  int*    lists  = counts + 64;                                    // [E][N]

  hipMemsetAsync(counts, 0, 64 * sizeof(int), stream);

  cvt_wT_kernel<<<dim3(512, E_EXP), 256, 0, stream>>>(W1, W1t, D_DIM, H_DIM);
  cvt_wT_kernel<<<dim3(512, E_EXP), 256, 0, stream>>>(W2, W2t, H_DIM, D_DIM);
  cvt_x_kernel<<<1024, 256, 0, stream>>>(x, Xb, N_TOK * D_DIM);

  router_kernel<<<N_TOK / 4, 128, 0, stream>>>(x, Wp, bp, probs, chosen, counts, lists);

  const size_t smemBytes = (size_t)2 * TILE_M * D_DIM * sizeof(__bf16);  // 128 KB
  hipFuncSetAttribute((const void*)moe_ffn_kernel,
                      hipFuncAttributeMaxDynamicSharedMemorySize, (int)smemBytes);
  moe_ffn_kernel<<<dim3(N_TOK / TILE_M, E_EXP), 256, smemBytes, stream>>>(
      Xb, W1t, W2t, b1, b2, counts, lists, out);
}
